// EdgeAwareRelationAttention_26036091748725
// MI455X (gfx1250) — matrix operand order repaired and checked by reading, not compiled, not run on hardware
//
#include <hip/hip_runtime.h>
#include <hip/hip_bf16.h>

#define TOKEN_DIM 256
#define HEADS     8
#define HEAD_DIM  32
#define EDGE_DIM  16
#define EDGE_EMB  16
#define BATCH     4
#define SEQ       2048
#define KNBR      32
#define ATTN_ROW  (2 * HEAD_DIM + EDGE_EMB)   // 80

typedef __attribute__((ext_vector_type(16))) _Float16 v16h;
typedef __attribute__((ext_vector_type(8)))  _Float16 v8h;
typedef __attribute__((ext_vector_type(8)))  float    v8f;
typedef __attribute__((ext_vector_type(4)))  float    v4f;

// ---------------------------------------------------------------------------
// fp32 -> f16 bulk convert (8 elements / thread, b128 in / b128 out)
// ---------------------------------------------------------------------------
__global__ __launch_bounds__(256)
void cvt_f32_to_f16(const float* __restrict__ in, _Float16* __restrict__ out, int n)
{
    int i = (blockIdx.x * blockDim.x + threadIdx.x) * 8;
    if (i >= n) return;
    v4f x0 = *(const v4f*)(in + i);
    v4f x1 = *(const v4f*)(in + i + 4);
    v8h y;
#pragma unroll
    for (int j = 0; j < 4; ++j) {
        y[j]     = (_Float16)x0[j];
        y[4 + j] = (_Float16)x1[j];
    }
    *(v8h*)(out + i) = y;
}

// ---------------------------------------------------------------------------
// Y[M,256](f16) = X[M,256](f16) @ W[256,256](f16)^T  via v_wmma_f32_16x16x32_f16
// One 16(M) x 32(N) tile per wave: A fragment reused by 2 WMMAs per k-step,
// K=256 fully unrolled -> 16 WMMAs / wave, pure b128 loads (no converts).
//   A-frag: lane l -> row l&15, contiguous K {(l>>4)*16 .. +15}
//   B-frag: lane l -> W row (out col) l&15, contiguous K {(l>>4)*16 .. +15}
//   C-frag: VGPR r -> row (l>>4)*8+r, col l&15
// ---------------------------------------------------------------------------
__global__ __launch_bounds__(256)
void gemm256_wmma_f16(const _Float16* __restrict__ X, const _Float16* __restrict__ W,
                      _Float16* __restrict__ Y, int M)
{
    const int lane = threadIdx.x & 31;
    const int wave = threadIdx.x >> 5;
    const int tile = blockIdx.x * 8 + wave;
    const int NT2  = TOKEN_DIM / 32;             // 8 N-supertile columns
    const int tm   = tile / NT2;
    const int tn2  = tile % NT2;                 // n0 = tn2*32
    if (tm * 16 >= M) return;

    const int l15 = lane & 15;
    const int hi  = lane >> 4;

    const _Float16* aptr  = X + (size_t)(tm * 16 + l15) * TOKEN_DIM + hi * 16;
    const _Float16* b0ptr = W + (size_t)(tn2 * 32 + l15) * TOKEN_DIM + hi * 16;
    const _Float16* b1ptr = b0ptr + (size_t)16 * TOKEN_DIM;

    v8f c0 = {}, c1 = {};
#pragma unroll
    for (int kk = 0; kk < TOKEN_DIM; kk += 32) {
        v8h a_lo = *(const v8h*)(aptr + kk);          // K base..base+7
        v8h a_hi = *(const v8h*)(aptr + kk + 8);      // K base+8..base+15
        v16h A = __builtin_shufflevector(a_lo, a_hi,
                                         0, 1, 2, 3, 4, 5, 6, 7,
                                         8, 9, 10, 11, 12, 13, 14, 15);
        v16h B0 = *(const v16h*)(b0ptr + kk);         // 16 contiguous K
        v16h B1 = *(const v16h*)(b1ptr + kk);

        c0 = __builtin_amdgcn_wmma_f32_16x16x32_f16(false, A, false, B0,
                                                    (short)0, c0, false, false);
        c1 = __builtin_amdgcn_wmma_f32_16x16x32_f16(false, A, false, B1,
                                                    (short)0, c1, false, false);
    }

    _Float16* o = Y + (size_t)(tm * 16 + hi * 8) * TOKEN_DIM + tn2 * 32 + l15;
#pragma unroll
    for (int r = 0; r < 8; ++r) {
        o[(size_t)r * TOKEN_DIM]      = (_Float16)c0[r];
        o[(size_t)r * TOKEN_DIM + 16] = (_Float16)c1[r];
    }
}

// ---------------------------------------------------------------------------
// out[row,h] = sum_d P[row, h*32+d] * attn[h, aoff+d]   (per-row head scores)
// ---------------------------------------------------------------------------
__global__ __launch_bounds__(256)
void head_dot(const _Float16* __restrict__ P, const float* __restrict__ attn,
              int aoff, float* __restrict__ out, int rows)
{
    int id = blockIdx.x * blockDim.x + threadIdx.x;
    if (id >= rows * HEADS) return;
    int row = id >> 3;
    int h   = id & 7;
    const _Float16* p = P + (size_t)row * TOKEN_DIM + h * HEAD_DIM;
    const float*    a = attn + h * ATTN_ROW + aoff;
    float s = 0.f;
#pragma unroll
    for (int d = 0; d < HEAD_DIM; ++d) s = fmaf((float)p[d], a[d], s);
    out[id] = s;
}

// ---------------------------------------------------------------------------
// s_edge[s,k,h] = sum_e ( edge_attr[s,k,:] . W_edge[h*16+e,:] ) * attn[h,64+e]
// ---------------------------------------------------------------------------
__global__ __launch_bounds__(256)
void edge_score(const float* __restrict__ edge_attr, const float* __restrict__ W_edge,
                const float* __restrict__ attn, float* __restrict__ s_edge)
{
    int id = blockIdx.x * blockDim.x + threadIdx.x;   // (s*K + k)*8 + h
    if (id >= SEQ * KNBR * HEADS) return;
    int h  = id & 7;
    int sk = id >> 3;

    const float* e = edge_attr + (size_t)sk * EDGE_DIM;
    float ev[EDGE_DIM];
#pragma unroll
    for (int f = 0; f < EDGE_DIM; ++f) ev[f] = e[f];

    const float* ae = attn + h * ATTN_ROW + 2 * HEAD_DIM;
    float s = 0.f;
#pragma unroll
    for (int q = 0; q < EDGE_EMB; ++q) {
        const float* w = W_edge + (size_t)(h * EDGE_EMB + q) * EDGE_DIM;
        float d = 0.f;
#pragma unroll
        for (int f = 0; f < EDGE_DIM; ++f) d = fmaf(ev[f], w[f], d);
        s = fmaf(d, ae[q], s);
    }
    s_edge[id] = s;
}

// ---------------------------------------------------------------------------
// One wave32 per (b,s):
//   lane = k: prefetch neighbor row (global_prefetch_b8), fused score ->
//             leaky_relu -> wave-shuffle softmax over K=32 -> LDS
//   e_acc[h,f] = sum_k alpha[k,h] * edge_attr[s,k,f]           (factored W_ev)
//   lane = d%32: msg[h] = sum_k alpha[k,h]*src_p[b,nbr_k,h*32+lane]
//                 + sum_f e_acc[h,f]*W_ev[h*32+lane, f]
// ---------------------------------------------------------------------------
__global__ __launch_bounds__(256)
void attn_message(const _Float16* __restrict__ src_p, const float* __restrict__ s_srcv,
                  const float* __restrict__ s_dstv,  const float* __restrict__ s_edgev,
                  const int*   __restrict__ nbr_idx, const float* __restrict__ edge_attr,
                  const float* __restrict__ W_ev,    float* __restrict__ out)
{
    __shared__ float sh_alpha[8][KNBR][HEADS];
    __shared__ int   sh_nbr[8][KNBR];
    __shared__ float sh_edge[8][KNBR][EDGE_DIM];
    __shared__ float sh_eacc[8][HEADS][EDGE_DIM];

    const int lane = threadIdx.x & 31;
    const int wave = threadIdx.x >> 5;
    const int task = blockIdx.x * 8 + wave;      // b*SEQ + s  (grid covers B*S exactly)
    const int b    = task / SEQ;
    const int s    = task % SEQ;

    // ---- phase 0: lane = neighbor k -------------------------------------
    const int k   = lane;
    const int nbr = nbr_idx[s * KNBR + k];
    sh_nbr[wave][k] = nbr;

    // warm the gather working set: each lane prefetches its neighbor's
    // full 512B f16 row (4 cachelines) -> 16KB per (b,s) resident before phase 2
    const _Float16* prow = src_p + ((size_t)b * SEQ + nbr) * TOKEN_DIM;
    __builtin_prefetch(prow,       0, 3);
    __builtin_prefetch(prow + 64,  0, 3);
    __builtin_prefetch(prow + 128, 0, 3);
    __builtin_prefetch(prow + 192, 0, 3);

    const float* erow = edge_attr + (size_t)(s * KNBR + k) * EDGE_DIM;
#pragma unroll
    for (int f = 0; f < EDGE_DIM; f += 4) {
        v4f t = *(const v4f*)(erow + f);
        sh_edge[wave][k][f + 0] = t[0];
        sh_edge[wave][k][f + 1] = t[1];
        sh_edge[wave][k][f + 2] = t[2];
        sh_edge[wave][k][f + 3] = t[3];
    }

    const float* pd = s_dstv + (size_t)task * HEADS;
    const float* ps = s_srcv + ((size_t)b * SEQ + nbr) * HEADS;
    const float* pe = s_edgev + (size_t)(s * KNBR + k) * HEADS;

    float sc[HEADS];
#pragma unroll
    for (int h = 0; h < HEADS; ++h) {
        float v = pd[h] + ps[h] + pe[h];
        sc[h] = v > 0.f ? v : 0.2f * v;          // leaky_relu(0.2)
    }

    // softmax across the 32 lanes (= K neighbors), independently per head
#pragma unroll
    for (int h = 0; h < HEADS; ++h) {
        float m = sc[h];
#pragma unroll
        for (int off = 16; off; off >>= 1) m = fmaxf(m, __shfl_xor(m, off, 32));
        float e = __expf(sc[h] - m);
        float t = e;
#pragma unroll
        for (int off = 16; off; off >>= 1) t += __shfl_xor(t, off, 32);
        sh_alpha[wave][k][h] = e / t;
    }
    __syncthreads();

    // ---- phase 1: e_acc[h,f] = sum_k alpha[k,h] * edge[k,f]  (4 per lane) --
#pragma unroll
    for (int t = 0; t < 4; ++t) {
        int id2 = lane * 4 + t;
        int h   = id2 >> 4;
        int f   = id2 & 15;
        float acc = 0.f;
#pragma unroll 8
        for (int kk = 0; kk < KNBR; ++kk)
            acc = fmaf(sh_alpha[wave][kk][h], sh_edge[wave][kk][f], acc);
        sh_eacc[wave][h][f] = acc;
    }
    __syncthreads();

    // ---- phase 2: alpha-weighted gather of src_p (lane = channel-in-head) --
    float msg[HEADS] = {};
    for (int kk = 0; kk < KNBR; ++kk) {
        const int nb = sh_nbr[wave][kk];
        const _Float16* g = src_p + ((size_t)b * SEQ + nb) * TOKEN_DIM + lane;
        float al[HEADS];
#pragma unroll
        for (int j = 0; j < HEADS; ++j) al[j] = sh_alpha[wave][kk][j];
#pragma unroll
        for (int j = 0; j < HEADS; ++j)
            msg[j] = fmaf(al[j], (float)g[j * HEAD_DIM], msg[j]);  // coalesced
    }

    // ---- phase 3: factored edge-value term + store ------------------------
#pragma unroll
    for (int j = 0; j < HEADS; ++j) {
        const float* w = W_ev + (size_t)(j * HEAD_DIM + lane) * EDGE_DIM;
        float acc = 0.f;
#pragma unroll
        for (int f = 0; f < EDGE_DIM; ++f)
            acc = fmaf(sh_eacc[wave][j][f], w[f], acc);
        msg[j] += acc;
    }

    float* o = out + (size_t)task * TOKEN_DIM + lane;
#pragma unroll
    for (int j = 0; j < HEADS; ++j)
        o[j * HEAD_DIM] = msg[j];
}

// ---------------------------------------------------------------------------
extern "C" void kernel_launch(void* const* d_in, const int* in_sizes, int n_in,
                              void* d_out, int out_size, void* d_ws, size_t ws_size,
                              hipStream_t stream)
{
    const float* src_tok = (const float*)d_in[0];
    const float* tgt_tok = (const float*)d_in[1];
    const int*   nbr     = (const int*)  d_in[2];
    const float* edge    = (const float*)d_in[3];
    const float* W_src   = (const float*)d_in[4];
    const float* W_dst   = (const float*)d_in[5];
    const float* W_edge  = (const float*)d_in[6];
    const float* W_ev    = (const float*)d_in[7];
    const float* attn    = (const float*)d_in[8];
    float* out = (float*)d_out;

    const int M  = BATCH * SEQ;                  // 8192 rows
    const int MD = M * TOKEN_DIM;                // 2,097,152
    const int WD = TOKEN_DIM * TOKEN_DIM;        // 65,536

    // workspace carve-up (~19 MB)
    char* w = (char*)d_ws;
    _Float16* Xs_h  = (_Float16*)w; w += (size_t)MD * 2;
    _Float16* Xt_h  = (_Float16*)w; w += (size_t)MD * 2;
    _Float16* Ws_h  = (_Float16*)w; w += (size_t)WD * 2;
    _Float16* Wd_h  = (_Float16*)w; w += (size_t)WD * 2;
    _Float16* src_p = (_Float16*)w; w += (size_t)MD * 2;
    _Float16* dst_p = (_Float16*)w; w += (size_t)MD * 2;
    float* s_srcv   = (float*)w;    w += (size_t)M * HEADS * 4;
    float* s_dstv   = (float*)w;    w += (size_t)M * HEADS * 4;
    float* s_edgev  = (float*)w;

    // f16 staging (one-time converts; removes cvt from WMMA hot loop)
    cvt_f32_to_f16<<<MD / (256 * 8), 256, 0, stream>>>(src_tok, Xs_h, MD);
    cvt_f32_to_f16<<<MD / (256 * 8), 256, 0, stream>>>(tgt_tok, Xt_h, MD);
    cvt_f32_to_f16<<<WD / (256 * 8), 256, 0, stream>>>(W_src,   Ws_h, WD);
    cvt_f32_to_f16<<<WD / (256 * 8), 256, 0, stream>>>(W_dst,   Wd_h, WD);

    // 512(M-tiles) x 8(N-supertiles) = 4096 wave-tiles, 8 waves/block
    gemm256_wmma_f16<<<512, 256, 0, stream>>>(Xs_h, Ws_h, src_p, M);
    gemm256_wmma_f16<<<512, 256, 0, stream>>>(Xt_h, Wd_h, dst_p, M);

    head_dot<<<(M * HEADS) / 256, 256, 0, stream>>>(src_p, attn, HEAD_DIM, s_srcv, M);
    head_dot<<<(M * HEADS) / 256, 256, 0, stream>>>(dst_p, attn, 0,        s_dstv, M);

    edge_score<<<(SEQ * KNBR * HEADS) / 256, 256, 0, stream>>>(edge, W_edge, attn, s_edgev);

    attn_message<<<M / 8, 256, 0, stream>>>(src_p, s_srcv, s_dstv, s_edgev,
                                            nbr, edge, W_ev, out);
}
